// Emb_ATT_Layers_18279380811820
// MI455X (gfx1250) — compile-verified, matrix operand order checked by reading, not run on hardware
//
#include <hip/hip_runtime.h>
#include <hip/hip_bf16.h>

// ---------------------------------------------------------------------------
// Problem constants (match reference)
// ---------------------------------------------------------------------------
#define LL   4
#define NN   50000
#define EMB  128
#define HH   4
#define HD   32
#define RR   16
#define HID  64
#define LBL  32
#define EE   625000

typedef __attribute__((ext_vector_type(16))) __bf16   v16bf;
typedef __attribute__((ext_vector_type(8)))  float    v8f;
typedef __attribute__((ext_vector_type(4)))  unsigned uint4s;
typedef __attribute__((ext_vector_type(8)))  unsigned uint8s;

// ---------------------------------------------------------------------------
// Tile load for the CDNA5 16-bit WMMA operand layout.
// Lane l: row = l&15, hi = l>>4.
//   elements 0..7  hold K = hi*8 + 0..7        (contiguous 16B)
//   elements 8..15 hold K = 16 + hi*8 + 0..7   (contiguous 16B)
// => two b128 loads per operand per k-step (global or ds).
// ---------------------------------------------------------------------------
__device__ __forceinline__ v16bf load_tile16(const __bf16* p) {
    v16bf r;
    uint4* u = reinterpret_cast<uint4*>(&r);
    u[0] = *reinterpret_cast<const uint4*>(p);
    u[1] = *reinterpret_cast<const uint4*>(p + 16);
    return r;
}

// ---------------------------------------------------------------------------
// bf16 GEMM with TDM-staged A panel, compile-time K, internal batch loop,
// 2-wide N-tile unroll (all Nc here are multiples of 32).
//   grid.x = ceil(tilesM/8); block = 256 threads (8 waves).
//   Block stages a 128-row A panel into LDS with one tensor_load_to_lds;
//   wave w owns M-tile (tmBase+w), hoists its A rows into registers from LDS
//   once, then for every batch b and N-tile pair:
//       C[b] (MxNc,f32) = A (MxKT) * Bt[b]^T (+bias)
//   Bt is (Nc x KT) row-major bf16.  All WMMA control flow is wave-uniform.
// ---------------------------------------------------------------------------
template <int KT, bool HASBIAS, bool HASCBF>
__global__ void gemm_bf16_tdm_kernel(const __bf16* __restrict__ A,
                                     const __bf16* __restrict__ Bt,
                                     const float*  __restrict__ bias,
                                     float*        __restrict__ C,
                                     __bf16*       __restrict__ Cbf,
                                     int M, int Nc, int nbatch,
                                     long bStrideBt, long bStrideC) {
    constexpr int KSTEPS = KT / 32;
    extern __shared__ __bf16 smem[];                  // 8*16*KT bf16
    const int wave   = threadIdx.x >> 5;
    const int lane   = threadIdx.x & 31;
    const int tilesN = Nc >> 4;
    const int tilesM = M  >> 4;
    const int tmBase = blockIdx.x * 8;

    // ---- stage A panel: rows [tmBase*16, tmBase*16+128) x KT, contiguous ----
#if defined(__HIP_DEVICE_COMPILE__)
    if (wave == 0) {
        const unsigned long long gbase =
            (unsigned long long)(const void*)A +
            (unsigned long long)tmBase * 16 * KT * 2;
        const unsigned ldsaddr = (unsigned)(unsigned long long)(void*)smem;
        long long remain = (long long)M * KT - (long long)tmBase * 16 * KT;
        const unsigned td0   = (unsigned)remain;       // OOB rows -> zero-fill
        const unsigned tile0 = (unsigned)(128 * KT);   // elements (<= 16384)

        uint4s g0; uint8s g1;
        g0[0] = 1u;                                    // count=1, user mode
        g0[1] = ldsaddr;                               // lds_addr
        g0[2] = (unsigned)(gbase & 0xffffffffu);       // global_addr[31:0]
        g0[3] = (unsigned)((gbase >> 32) & 0x01ffffffu)
              | (2u << 30);                            // addr[56:32] | type=2
        g1[0] = 1u << 16;                              // data_size=1 (2 bytes)
        g1[1] = (td0 & 0xffffu) << 16;                 // tensor_dim0[15:0]
        g1[2] = (td0 >> 16) | (1u << 16);              // td0[31:16] | td1=1 lo
        g1[3] = tile0 << 16;                           // td1 hi=0 | tile_dim0
        g1[4] = 0u;                                    // tile_dim1/2 unused
        g1[5] = td0;                                   // dim0_stride lo
        g1[6] = 0u;                                    // dim0_stride hi
        g1[7] = 0u;
        asm volatile("tensor_load_to_lds %0, %1" :: "s"(g0), "s"(g1) : "memory");
        __builtin_amdgcn_s_wait_tensorcnt(0);
    }
#endif
    __syncthreads();

    const int tm = tmBase + wave;
    if (tm >= tilesM) return;                          // wave-uniform exit

    const int m  = lane & 15;                          // row of A / col of C
    const int hi = lane >> 4;

    // hoist this wave's A rows from LDS into registers (KSTEPS k-steps)
    const __bf16* As = smem + (wave * 16 + m) * KT + hi * 8;
    v16bf areg[KSTEPS];
#pragma unroll
    for (int kk = 0; kk < KSTEPS; ++kk)
        areg[kk] = load_tile16(As + kk * 32);

    for (int bb = 0; bb < nbatch; ++bb) {
        const __bf16* Bbase = Bt + (long)bb * bStrideBt + (long)m * KT + hi * 8;
        float*  Cp  = C + (long)bb * bStrideC;
        __bf16* Cbp = HASCBF ? (Cbf + (long)bb * bStrideC) : nullptr;

        for (int tn = 0; tn < tilesN; tn += 2) {
            const __bf16* Bb0 = Bbase + (long)tn * 16 * KT;
            const __bf16* Bb1 = Bb0 + 16 * KT;
            v8f c0 = {}, c1 = {};
#pragma unroll
            for (int kk = 0; kk < KSTEPS; ++kk) {
                v16bf bv0 = load_tile16(Bb0 + kk * 32);
                v16bf bv1 = load_tile16(Bb1 + kk * 32);
                c0 = __builtin_amdgcn_wmma_f32_16x16x32_bf16(
                         false, areg[kk], false, bv0, (short)0, c0, false, false);
                c1 = __builtin_amdgcn_wmma_f32_16x16x32_bf16(
                         false, areg[kk], false, bv1, (short)0, c1, false, false);
            }
            float b0 = 0.0f, b1 = 0.0f;
            if (HASBIAS) {
                b0 = bias[tn * 16 + m];
                b1 = bias[tn * 16 + 16 + m];
            }
            // row base: C row (tm*16 + hi*8 + j), col (tn*16 + m)
            const long rbase = (long)(tm * 16 + hi * 8) * Nc + tn * 16 + m;
            float* Crow = Cp + rbase;
#pragma unroll
            for (int j = 0; j < 8; ++j) {
                const float v0 = c0[j] + b0;
                const float v1 = c1[j] + b1;
                Crow[(long)j * Nc]      = v0;
                Crow[(long)j * Nc + 16] = v1;
                if (HASCBF) {
                    __bf16* Brow = Cbp + rbase;
                    Brow[(long)j * Nc]      = (__bf16)v0;
                    Brow[(long)j * Nc + 16] = (__bf16)v1;
                }
            }
        }
    }
}

// ---------------------------------------------------------------------------
// f32 -> bf16 (optional relu)
// ---------------------------------------------------------------------------
__global__ void cvt_bf16_kernel(const float* __restrict__ in,
                                __bf16* __restrict__ out, long n, int relu) {
    long i = (long)blockIdx.x * blockDim.x + threadIdx.x;
    if (i >= n) return;
    float v = in[i];
    if (relu) v = fmaxf(v, 0.0f);
    out[i] = (__bf16)v;
}

// out[b][n*K + k] = (bf16) in[b][k*Nc + n]   (build Bt from a (K,Nc) weight)
__global__ void transpose_cvt_kernel(const float* __restrict__ in,
                                     __bf16* __restrict__ out,
                                     int K, int Nc,
                                     long inBatchStride, long outBatchStride) {
    int t = blockIdx.x * blockDim.x + threadIdx.x;
    if (t >= K * Nc) return;
    int n = t / K;
    int k = t - n * K;
    out[(long)blockIdx.y * outBatchStride + t] =
        (__bf16)in[(long)blockIdx.y * inBatchStride + (long)k * Nc + n];
}

__global__ void zero_kernel(float* __restrict__ p, long n) {
    long i = (long)blockIdx.x * blockDim.x + threadIdx.x;
    if (i < n) p[i] = 0.0f;
}

// ---------------------------------------------------------------------------
// Attention for query row l=0 only (all the reference consumes).
// One thread per (node, head).  Writes bf16 (next GEMM input).
// ---------------------------------------------------------------------------
__global__ void attn_kernel(const float* __restrict__ qkv,
                            __bf16* __restrict__ obf) {
    int t = blockIdx.x * blockDim.x + threadIdx.x;
    if (t >= NN * HH) return;
    const int n = t >> 2;
    const int h = t & 3;

    float4 qv[8];
    const float* qp = qkv + (long)n * 384 + h * HD;
#pragma unroll
    for (int i = 0; i < 8; ++i) qv[i] = *(const float4*)(qp + 4 * i);

    float s[4];
#pragma unroll
    for (int m2 = 0; m2 < 4; ++m2) {
        const float* kp = qkv + ((long)m2 * NN + n) * 384 + 128 + h * HD;
        float acc = 0.0f;
#pragma unroll
        for (int i = 0; i < 8; ++i) {
            float4 kk = *(const float4*)(kp + 4 * i);
            acc += qv[i].x * kk.x + qv[i].y * kk.y +
                   qv[i].z * kk.z + qv[i].w * kk.w;
        }
        s[m2] = acc * 0.17677669529663687f;   // 1/sqrt(32)
    }
    float mx = fmaxf(fmaxf(s[0], s[1]), fmaxf(s[2], s[3]));
    float e[4], sum = 0.0f;
#pragma unroll
    for (int m2 = 0; m2 < 4; ++m2) { e[m2] = __expf(s[m2] - mx); sum += e[m2]; }
    const float inv = 1.0f / sum;

    float4 ov[8];
#pragma unroll
    for (int i = 0; i < 8; ++i) ov[i] = make_float4(0.f, 0.f, 0.f, 0.f);
#pragma unroll
    for (int m2 = 0; m2 < 4; ++m2) {
        const float w = e[m2] * inv;
        const float* vp = qkv + ((long)m2 * NN + n) * 384 + 256 + h * HD;
#pragma unroll
        for (int i = 0; i < 8; ++i) {
            float4 vv = *(const float4*)(vp + 4 * i);
            ov[i].x += w * vv.x; ov[i].y += w * vv.y;
            ov[i].z += w * vv.z; ov[i].w += w * vv.w;
        }
    }
    __bf16* op = obf + (long)n * EMB + h * HD;
#pragma unroll
    for (int i = 0; i < 8; ++i) {
        op[4 * i + 0] = (__bf16)ov[i].x;
        op[4 * i + 1] = (__bf16)ov[i].y;
        op[4 * i + 2] = (__bf16)ov[i].z;
        op[4 * i + 3] = (__bf16)ov[i].w;
    }
}

// ---------------------------------------------------------------------------
// Per-(relation,dst) edge counts for the mean normalization.
// ---------------------------------------------------------------------------
__global__ void count_kernel(const long long* __restrict__ ei,
                             const long long* __restrict__ et,
                             float* __restrict__ cnt) {
    int e = blockIdx.x * blockDim.x + threadIdx.x;
    if (e >= EE) return;
    __builtin_prefetch(ei + EE + e + 2048, 0, 0);
    const int d = (int)ei[EE + e];
    const int r = (int)et[e];
    atomicAdd(&cnt[(long)r * NN + d], 1.0f);
}

// ---------------------------------------------------------------------------
// Edge scatter:  acc[dst][j] += msg[etype][src][j] / max(cnt[etype,dst],1)
// (transform-first mean aggregation).  One thread per (edge, feature).
// ---------------------------------------------------------------------------
__global__ void scatter_kernel(const long long* __restrict__ ei,
                               const long long* __restrict__ et,
                               const float* __restrict__ cnt,
                               const float* __restrict__ msg,
                               float* __restrict__ acc,
                               int F, int fshift) {
    long t = (long)blockIdx.x * blockDim.x + threadIdx.x;
    if (t >= (long)EE * F) return;
    const int  j = (int)(t & (F - 1));
    const long e = t >> fshift;
    __builtin_prefetch(ei + e + 1024, 0, 0);
    const int  s = (int)ei[e];
    const int  d = (int)ei[EE + e];
    const int  r = (int)et[e];
    const float c = cnt[(long)r * NN + d];
    const float w = 1.0f / fmaxf(c, 1.0f);
    atomicAdd(&acc[(long)d * F + j], msg[((long)r * NN + s) * F + j] * w);
}

// ---------------------------------------------------------------------------
// Row softmax over 32 labels: one wave32 per node, shfl_xor reductions.
// ---------------------------------------------------------------------------
__global__ void softmax_kernel(const float* __restrict__ in,
                               float* __restrict__ out) {
    const int wid  = (int)((blockIdx.x * (long)blockDim.x + threadIdx.x) >> 5);
    const int lane = threadIdx.x & 31;
    if (wid >= NN) return;                          // wave-uniform exit
    float v = in[(long)wid * LBL + lane];
    float m = v;
#pragma unroll
    for (int s = 16; s > 0; s >>= 1) m = fmaxf(m, __shfl_xor(m, s, 32));
    const float e = __expf(v - m);
    float sum = e;
#pragma unroll
    for (int s = 16; s > 0; s >>= 1) sum += __shfl_xor(sum, s, 32);
    out[(long)wid * LBL + lane] = e / sum;
}

// ---------------------------------------------------------------------------
// Host driver
// ---------------------------------------------------------------------------
static inline int divup(long a, long b) { return (int)((a + b - 1) / b); }

extern "C" void kernel_launch(void* const* d_in, const int* in_sizes, int n_in,
                              void* d_out, int out_size, void* d_ws, size_t ws_size,
                              hipStream_t stream) {
    const float*     emb = (const float*)d_in[0];
    const float*     wi  = (const float*)d_in[1];
    const float*     bi  = (const float*)d_in[2];
    const float*     wo  = (const float*)d_in[3];
    const float*     bo  = (const float*)d_in[4];
    const float*     w1  = (const float*)d_in[5];
    const float*     r1  = (const float*)d_in[6];
    const float*     b1  = (const float*)d_in[7];
    const float*     w2  = (const float*)d_in[8];
    const float*     r2  = (const float*)d_in[9];
    const float*     b2  = (const float*)d_in[10];
    const long long* ei  = (const long long*)d_in[11];
    const long long* et  = (const long long*)d_in[12];
    float* out = (float*)d_out;
    char*  ws  = (char*)d_ws;

    // ---- workspace layout (256B aligned); xW1/xW2 alias the dead emb/qkv ----
    size_t off = 0;
    auto alloc = [&](size_t bytes) {
        size_t o = off; off += (bytes + 255) & ~(size_t)255; return o;
    };
    const size_t o_embbf = alloc((size_t)LL * NN * EMB * 2);   // 51.2 MB
    const size_t o_qkv   = alloc((size_t)LL * NN * 384 * 4);   // 307.2 MB
    const size_t o_wibf  = alloc((size_t)384 * 128 * 2);
    const size_t o_wobf  = alloc((size_t)128 * 128 * 2);
    const size_t o_r1t   = alloc((size_t)HID * EMB * 2);
    const size_t o_w1t   = alloc((size_t)RR * HID * EMB * 2);
    const size_t o_r2t   = alloc((size_t)LBL * HID * 2);
    const size_t o_w2t   = alloc((size_t)RR * LBL * HID * 2);
    const size_t o_obf   = alloc((size_t)NN * EMB * 2);
    const size_t o_xatt  = alloc((size_t)NN * EMB * 4);
    const size_t o_xattb = alloc((size_t)NN * EMB * 2);
    const size_t o_cnt   = alloc((size_t)RR * NN * 4);
    const size_t o_h1    = alloc((size_t)NN * HID * 4);
    const size_t o_h1bf  = alloc((size_t)NN * HID * 2);
    const size_t o_out2  = alloc((size_t)NN * LBL * 4);
    // aliases over [o_embbf, ...) which are dead once attention is done
    const size_t o_xW1 = o_embbf;                                // 204.8 MB
    const size_t o_xW2 = o_embbf + (size_t)RR * NN * HID * 4;    // 102.4 MB
    (void)ws_size; (void)in_sizes; (void)n_in; (void)out_size;

    __bf16* emb_bf = (__bf16*)(ws + o_embbf);
    float*  qkv    = (float*) (ws + o_qkv);
    __bf16* wi_bf  = (__bf16*)(ws + o_wibf);
    __bf16* wo_bf  = (__bf16*)(ws + o_wobf);
    __bf16* r1t    = (__bf16*)(ws + o_r1t);
    __bf16* w1t    = (__bf16*)(ws + o_w1t);
    __bf16* r2t    = (__bf16*)(ws + o_r2t);
    __bf16* w2t    = (__bf16*)(ws + o_w2t);
    __bf16* obf    = (__bf16*)(ws + o_obf);
    float*  xatt   = (float*) (ws + o_xatt);
    __bf16* xattb  = (__bf16*)(ws + o_xattb);
    float*  cnt    = (float*) (ws + o_cnt);
    float*  h1acc  = (float*) (ws + o_h1);
    __bf16* h1bf   = (__bf16*)(ws + o_h1bf);
    float*  out2   = (float*) (ws + o_out2);
    float*  xW1    = (float*) (ws + o_xW1);
    float*  xW2    = (float*) (ws + o_xW2);

    const int B = 256;

    // ---- weight / activation prep (bf16) ----
    cvt_bf16_kernel<<<divup((long)LL * NN * EMB, B), B, 0, stream>>>(
        emb, emb_bf, (long)LL * NN * EMB, 0);
    cvt_bf16_kernel<<<divup(384 * 128, B), B, 0, stream>>>(wi, wi_bf, 384 * 128, 0);
    cvt_bf16_kernel<<<divup(128 * 128, B), B, 0, stream>>>(wo, wo_bf, 128 * 128, 0);
    transpose_cvt_kernel<<<divup(EMB * HID, B), B, 0, stream>>>(
        r1, r1t, EMB, HID, 0, 0);
    transpose_cvt_kernel<<<dim3(divup(EMB * HID, B), RR), B, 0, stream>>>(
        w1, w1t, EMB, HID, (long)EMB * HID, (long)EMB * HID);
    transpose_cvt_kernel<<<divup(HID * LBL, B), B, 0, stream>>>(
        r2, r2t, HID, LBL, 0, 0);
    transpose_cvt_kernel<<<dim3(divup(HID * LBL, B), RR), B, 0, stream>>>(
        w2, w2t, HID, LBL, (long)HID * LBL, (long)HID * LBL);

    // ---- qkv = emb @ Wi^T + bi : (200000x128)x(128x384), 12500 M-tiles ----
    gemm_bf16_tdm_kernel<EMB, true, false>
        <<<divup(LL * NN / 16, 8), B, 256 * EMB, stream>>>(
        emb_bf, wi_bf, bi, qkv, nullptr, LL * NN, 384, 1, 0, 0);

    // ---- attention (l = 0 only) ----
    attn_kernel<<<divup(NN * HH, B), B, 0, stream>>>(qkv, obf);

    // ---- x = attn_out0 @ Wo^T + bo ----
    gemm_bf16_tdm_kernel<EMB, true, true>
        <<<divup(NN / 16, 8), B, 256 * EMB, stream>>>(
        obf, wo_bf, bo, xatt, xattb, NN, EMB, 1, 0, 0);

    // ---- edge counts ----
    zero_kernel<<<divup((long)RR * NN, B), B, 0, stream>>>(cnt, (long)RR * NN);
    count_kernel<<<divup(EE, B), B, 0, stream>>>(ei, et, cnt);

    // ---- layer 1: xW1[r] = x @ W1[r] (R batches in-kernel), h1 = x@root1+b1 ----
    gemm_bf16_tdm_kernel<EMB, false, false>
        <<<divup(NN / 16, 8), B, 256 * EMB, stream>>>(
        xattb, w1t, nullptr, xW1, nullptr, NN, HID, RR,
        (long)EMB * HID, (long)NN * HID);
    gemm_bf16_tdm_kernel<EMB, true, false>
        <<<divup(NN / 16, 8), B, 256 * EMB, stream>>>(
        xattb, r1t, b1, h1acc, nullptr, NN, HID, 1, 0, 0);
    scatter_kernel<<<divup((long)EE * HID, B), B, 0, stream>>>(
        ei, et, cnt, xW1, h1acc, HID, 6);
    cvt_bf16_kernel<<<divup((long)NN * HID, B), B, 0, stream>>>(
        h1acc, h1bf, (long)NN * HID, /*relu=*/1);

    // ---- layer 2 ----
    gemm_bf16_tdm_kernel<HID, false, false>
        <<<divup(NN / 16, 8), B, 256 * HID, stream>>>(
        h1bf, w2t, nullptr, xW2, nullptr, NN, LBL, RR,
        (long)HID * LBL, (long)NN * LBL);
    gemm_bf16_tdm_kernel<HID, true, false>
        <<<divup(NN / 16, 8), B, 256 * HID, stream>>>(
        h1bf, r2t, b2, out2, nullptr, NN, LBL, 1, 0, 0);
    scatter_kernel<<<divup((long)EE * LBL, B), B, 0, stream>>>(
        ei, et, cnt, xW2, out2, LBL, 5);

    // ---- softmax -> d_out ----
    softmax_kernel<<<divup((long)NN * 32, B), B, 0, stream>>>(out2, out);
}
